// SimpleTransformer_15676630631066
// MI455X (gfx1250) — matrix operand order repaired and checked by reading, not compiled, not run on hardware
//
#include <hip/hip_runtime.h>
#include <math.h>

typedef __attribute__((ext_vector_type(2)))  float  v2f;
typedef __attribute__((ext_vector_type(8)))  float  v8f;
typedef __attribute__((ext_vector_type(8)))  __bf16 v8bf;
typedef __attribute__((ext_vector_type(16))) __bf16 v16bf;
typedef int v4i_ __attribute__((vector_size(4 * sizeof(int))));

#if __has_builtin(__builtin_amdgcn_global_load_async_to_lds_b128)
#define GEMM_ASYNC 1
#else
#define GEMM_ASYNC 0
#endif

__device__ __forceinline__ v8f wmma_f32(v2f a, v2f b, v8f c) {
  return __builtin_amdgcn_wmma_f32_16x16x4_f32(false, a, false, b, (short)0, c, false, false);
}
__device__ __forceinline__ v8f wmma_bf16(v16bf a, v16bf b, v8f c) {
  return __builtin_amdgcn_wmma_f32_16x16x32_bf16(false, a, false, b, (short)0, c, false, false);
}

// f32 -> bf16 bits, round-to-nearest-even
__device__ __forceinline__ unsigned bf16_bits_rne(float f) {
  unsigned u = __float_as_uint(f);
  return (u + 0x7FFFu + ((u >> 16) & 1u)) >> 16;
}
__device__ __forceinline__ float bf16_bits_to_f32(unsigned b) {
  return __uint_as_float(b << 16);
}
__device__ __forceinline__ __bf16 bits_to_bf16(unsigned b) {
  unsigned short s = (unsigned short)b;
  __bf16 r;
  __builtin_memcpy(&r, &s, 2);
  return r;
}
// split x into (hi, lo) bf16
__device__ __forceinline__ void split_bf16(float x, __bf16& hi, __bf16& lo) {
  unsigned hb = bf16_bits_rne(x);
  unsigned lb = bf16_bits_rne(x - bf16_bits_to_f32(hb));
  hi = bits_to_bf16(hb);
  lo = bits_to_bf16(lb);
}

// 16-byte global -> LDS copy (async on CDNA5 when available)
__device__ __forceinline__ void cp16(__bf16* dst_lds, const __bf16* src_g) {
#if GEMM_ASYNC
  __builtin_amdgcn_global_load_async_to_lds_b128(
      (__attribute__((address_space(1))) v4i_*)(const void*)src_g,
      (__attribute__((address_space(3))) v4i_*)(void*)dst_lds, 0, 0);
#else
  *(v8bf*)dst_lds = *(const v8bf*)src_g;
#endif
}
__device__ __forceinline__ void stage_fence() {
#if GEMM_ASYNC
#if __has_builtin(__builtin_amdgcn_s_wait_asynccnt)
  __builtin_amdgcn_s_wait_asynccnt(0);
#else
  asm volatile("s_wait_asynccnt 0x0" ::: "memory");
#endif
#endif
  __syncthreads();
}

// ---------------------------------------------------------------------------
// 1) xpe = x + positional encoding   (also emits bf16 hi/lo planes of xpe)
// ---------------------------------------------------------------------------
__global__ __launch_bounds__(256) void pe_add_kernel(const float* __restrict__ x,
                                                     float* __restrict__ xpe,
                                                     __bf16* __restrict__ xh,
                                                     __bf16* __restrict__ xl,
                                                     int S, int D, int total) {
  int i = blockIdx.x * blockDim.x + threadIdx.x;
  if (i >= total) return;
  int d = i % D;
  int s = (i / D) % S;
  int half2 = (d >> 1) << 1;
  const float neg_ln1e4_over_D = -9.210340371976184f / (float)D;
  float div = expf((float)half2 * neg_ln1e4_over_D);
  float ang = (float)s * div;
  float pe = (d & 1) ? cosf(ang) : sinf(ang);
  float val = x[i] + pe;
  xpe[i] = val;
  split_bf16(val, xh[i], xl[i]);
}

// ---------------------------------------------------------------------------
// 2) Weight prep: W[K,N] -> transposed hi/lo bf16 planes Wt[N][K]
// ---------------------------------------------------------------------------
__global__ __launch_bounds__(256) void split_transpose_kernel(const float* __restrict__ W,
                                                              __bf16* __restrict__ outH,
                                                              __bf16* __restrict__ outL,
                                                              int K, int N) {
  int i = blockIdx.x * blockDim.x + threadIdx.x;
  if (i >= K * N) return;
  int n = i / K;
  int k = i % K;
  float x = W[(size_t)k * N + n];
  split_bf16(x, outH[i], outL[i]);
}

// ---------------------------------------------------------------------------
// 3) Split-bf16 GEMM on pre-split operands:
//    C[M,N] = A[M,K] @ B[K,N] + bias  with A as hi/lo planes [M,K] and
//    B as transposed hi/lo planes [N,K].  Each K=32 tile:
//    Ahi*Bhi + Alo*Bhi + Ahi*Blo on v_wmma_f32_16x16x32_bf16 (f32 accum).
//    Double-buffered LDS, async global->LDS staging (ASYNCcnt).
//    Output: f32 (Cf) or hi/lo bf16 planes (Ch/Cl) for chained GEMMs.
// ---------------------------------------------------------------------------
#define BM 128
#define BN 64
#define BK 32
#define LDA 40  // padded bf16 row stride (80B: 16B-aligned chunks, de-banked)

__device__ __forceinline__ v16bf ld_afrag(const __bf16* row, int half) {
  // A fragment in contiguous k order (same as B): lane m holds 16 contiguous K starting at half*16
  v8bf a0 = *(const v8bf*)(row + half * 16);
  v8bf a1 = *(const v8bf*)(row + half * 16 + 8);
  return __builtin_shufflevector(a0, a1, 0, 1, 2, 3, 4, 5, 6, 7,
                                 8, 9, 10, 11, 12, 13, 14, 15);
}
__device__ __forceinline__ v16bf ld_bfrag(const __bf16* row, int half) {
  // B fragment in contiguous k order (same as A): lane n holds 16 contiguous K starting at half*16
  v8bf b0 = *(const v8bf*)(row + half * 16);
  v8bf b1 = *(const v8bf*)(row + half * 16 + 8);
  return __builtin_shufflevector(b0, b1, 0, 1, 2, 3, 4, 5, 6, 7,
                                 8, 9, 10, 11, 12, 13, 14, 15);
}

__global__ __launch_bounds__(256) void gemm_split_kernel(
    const __bf16* __restrict__ Ah, const __bf16* __restrict__ Al,   // [M,K]
    const __bf16* __restrict__ Bh, const __bf16* __restrict__ Bl,   // [N,K]
    const float* __restrict__ bias,
    float* __restrict__ Cf,                                          // f32 out (or null)
    __bf16* __restrict__ Ch, __bf16* __restrict__ Cl,                // split out (or null)
    int M, int K, int N, int relu) {
  __shared__ __bf16 As[2][2][BM][LDA];  // [buf][plane][m][k]
  __shared__ __bf16 Bs[2][2][BN][LDA];  // [buf][plane][n][k]

  const int tid  = threadIdx.x;
  const int m0   = blockIdx.x * BM;
  const int n0   = blockIdx.y * BN;
  const int wave = tid >> 5;
  const int lane = tid & 31;
  const int lm   = lane & 15;
  const int half = lane >> 4;
  const int rw   = wave * 16;

  v8f acc[4];
#pragma unroll
  for (int t = 0; t < 4; ++t)
#pragma unroll
    for (int j = 0; j < 8; ++j) acc[t][j] = 0.0f;

  auto stage = [&](int s, int k0) {
    // A: 2 planes * 128 rows * 4 chunks of 16B = 1024 chunks (4/thread)
#pragma unroll
    for (int i = tid; i < 1024; i += 256) {
      int plane = i >> 9;
      int r = (i >> 2) & 127;
      int c = (i & 3) << 3;
      const __bf16* src = (plane ? Al : Ah) + (size_t)(m0 + r) * K + k0 + c;
      cp16(&As[s][plane][r][c], src);
    }
    // B: 2 planes * 64 rows * 4 chunks = 512 chunks (2/thread)
#pragma unroll
    for (int i = tid; i < 512; i += 256) {
      int plane = i >> 8;
      int r = (i >> 2) & 63;
      int c = (i & 3) << 3;
      const __bf16* src = (plane ? Bl : Bh) + (size_t)(n0 + r) * K + k0 + c;
      cp16(&Bs[s][plane][r][c], src);
    }
  };

  int cur = 0;
  stage(0, 0);
  stage_fence();

  for (int k0 = 0; k0 < K; k0 += BK) {
    const int nxt = cur ^ 1;
    if (k0 + BK < K) stage(nxt, k0 + BK);

    const v16bf ah = ld_afrag(&As[cur][0][rw + lm][0], half);
    const v16bf al = ld_afrag(&As[cur][1][rw + lm][0], half);
#pragma unroll
    for (int nt = 0; nt < 4; ++nt) {
      const v16bf bh = ld_bfrag(&Bs[cur][0][nt * 16 + lm][0], half);
      const v16bf bl = ld_bfrag(&Bs[cur][1][nt * 16 + lm][0], half);
      acc[nt] = wmma_bf16(ah, bh, acc[nt]);
      acc[nt] = wmma_bf16(al, bh, acc[nt]);
      acc[nt] = wmma_bf16(ah, bl, acc[nt]);
    }
    stage_fence();  // staging of nxt done; all waves done reading cur
    cur = nxt;
  }

  // Store: C/D layout -> vgpr j is row (j + half*8), col = lane%16
#pragma unroll
  for (int nt = 0; nt < 4; ++nt) {
    int col = n0 + nt * 16 + lm;
    float bv = bias[col];
#pragma unroll
    for (int j = 0; j < 8; ++j) {
      int row = m0 + rw + j + (half << 3);
      float vv = acc[nt][j] + bv;
      if (relu) vv = fmaxf(vv, 0.0f);
      size_t idx = (size_t)row * N + col;
      if (Cf) {
        Cf[idx] = vv;
      } else {
        split_bf16(vv, Ch[idx], Cl[idx]);
      }
    }
  }
}

// ---------------------------------------------------------------------------
// 4) Flash attention: one wave per (b, h, 16-query block). dk=64, D=512.
//    Streaming softmax; scores never hit memory. Emits ctx as hi/lo planes.
// ---------------------------------------------------------------------------
__global__ __launch_bounds__(128) void flash_attn_kernel(const float* __restrict__ q,
                                                         const float* __restrict__ k,
                                                         const float* __restrict__ v,
                                                         __bf16* __restrict__ ctxh,
                                                         __bf16* __restrict__ ctxl,
                                                         int B, int S, int H) {
  const int D = 512, dk = 64;
  __shared__ float Pl[4][16 * 18];

  const int wave = threadIdx.x >> 5;
  const int lane = threadIdx.x & 31;
  const int gw   = blockIdx.x * 4 + wave;
  const int QB   = S / 16;
  const int qb   = gw % QB;
  const int bh   = gw / QB;
  const int h    = bh % H;
  const int b    = bh / H;

  const int lm = lane & 15;
  const int kh = (lane >> 4) << 1;
  float* P = &Pl[wave][0];

  const size_t qrow = ((size_t)b * S + (size_t)qb * 16 + lm) * D + (size_t)h * dk;
  v2f qf[16];
#pragma unroll
  for (int t = 0; t < 16; ++t) {
    qf[t].x = q[qrow + t * 4 + kh];
    qf[t].y = q[qrow + t * 4 + kh + 1];
  }

  float mrun[8], lrun[8];
  v8f acc[4];
#pragma unroll
  for (int j = 0; j < 8; ++j) { mrun[j] = -1e30f; lrun[j] = 0.0f; }
#pragma unroll
  for (int t = 0; t < 4; ++t)
#pragma unroll
    for (int j = 0; j < 8; ++j) acc[t][j] = 0.0f;

  const float scale = 0.125f;

  for (int kt = 0; kt < S / 16; ++kt) {
    const size_t kbase = ((size_t)b * S + (size_t)kt * 16) * D + (size_t)h * dk;

    v8f s;
#pragma unroll
    for (int j = 0; j < 8; ++j) s[j] = 0.0f;
#pragma unroll
    for (int t = 0; t < 16; ++t) {
      v2f bf;
      const size_t krow = kbase + (size_t)lm * D + t * 4 + kh;
      bf.x = k[krow];
      bf.y = k[krow + 1];
      s = wmma_f32(qf[t], bf, s);
    }

    float pj[8];
#pragma unroll
    for (int j = 0; j < 8; ++j) {
      float sv = s[j] * scale;
      float mx = sv;
      mx = fmaxf(mx, __shfl_xor(mx, 1, 32));
      mx = fmaxf(mx, __shfl_xor(mx, 2, 32));
      mx = fmaxf(mx, __shfl_xor(mx, 4, 32));
      mx = fmaxf(mx, __shfl_xor(mx, 8, 32));
      float mnew = fmaxf(mrun[j], mx);
      float p = expf(sv - mnew);
      float ls = p;
      ls += __shfl_xor(ls, 1, 32);
      ls += __shfl_xor(ls, 2, 32);
      ls += __shfl_xor(ls, 4, 32);
      ls += __shfl_xor(ls, 8, 32);
      float alpha = expf(mrun[j] - mnew);
      lrun[j] = lrun[j] * alpha + ls;
      mrun[j] = mnew;
#pragma unroll
      for (int nt = 0; nt < 4; ++nt) acc[nt][j] *= alpha;
      pj[j] = p;
    }

#pragma unroll
    for (int j = 0; j < 8; ++j) {
      int m = j + ((lane >> 4) << 3);
      P[m * 18 + lm] = pj[j];
    }
    __asm volatile("s_wait_dscnt 0" ::: "memory");

#pragma unroll
    for (int kk = 0; kk < 16; kk += 4) {
      v2f a;
      a.x = P[lm * 18 + kk + kh];
      a.y = P[lm * 18 + kk + kh + 1];
      const size_t vrow0 = ((size_t)b * S + (size_t)kt * 16 + kk + kh) * D + (size_t)h * dk;
#pragma unroll
      for (int nt = 0; nt < 4; ++nt) {
        v2f bf;
        bf.x = v[vrow0 + nt * 16 + lm];
        bf.y = v[vrow0 + D + nt * 16 + lm];
        acc[nt] = wmma_f32(a, bf, acc[nt]);
      }
    }
  }

#pragma unroll
  for (int nt = 0; nt < 4; ++nt)
#pragma unroll
    for (int j = 0; j < 8; ++j) {
      int m = j + ((lane >> 4) << 3);
      size_t row = (size_t)b * S + (size_t)qb * 16 + m;
      size_t idx = row * D + (size_t)h * dk + nt * 16 + lm;
      float o = acc[nt][j] / lrun[j];
      split_bf16(o, ctxh[idx], ctxl[idx]);
    }
}

// ---------------------------------------------------------------------------
// 5) out = LayerNorm(a + r)*g + beta; optionally also emits hi/lo planes
// ---------------------------------------------------------------------------
__global__ __launch_bounds__(256) void add_ln_kernel(const float* __restrict__ a,
                                                     const float* __restrict__ r,
                                                     const float* __restrict__ g,
                                                     const float* __restrict__ beta,
                                                     float* __restrict__ out,
                                                     __bf16* __restrict__ oh,
                                                     __bf16* __restrict__ ol) {
  const int D = 512;
  __shared__ float red[16];
  const size_t row = blockIdx.x;
  const int t = threadIdx.x;
  const float* pa = a + row * D;
  const float* pr = r + row * D;

  float x0 = pa[t] + pr[t];
  float x1 = pa[t + 256] + pr[t + 256];
  float sum = x0 + x1;
  float sq  = x0 * x0 + x1 * x1;
#pragma unroll
  for (int m = 1; m < 32; m <<= 1) {
    sum += __shfl_xor(sum, m, 32);
    sq  += __shfl_xor(sq,  m, 32);
  }
  const int wv = t >> 5, lane = t & 31;
  if (lane == 0) { red[wv] = sum; red[8 + wv] = sq; }
  __syncthreads();
  if (t == 0) {
    float s = 0.f, q = 0.f;
#pragma unroll
    for (int w = 0; w < 8; ++w) { s += red[w]; q += red[8 + w]; }
    red[0] = s;
    red[8] = q;
  }
  __syncthreads();
  const float mu  = red[0] / (float)D;
  const float var = red[8] / (float)D - mu * mu;
  const float inv = rsqrtf(var + 1e-5f);
  float y0 = (x0 - mu) * inv * g[t]       + beta[t];
  float y1 = (x1 - mu) * inv * g[t + 256] + beta[t + 256];
  out[row * D + t]       = y0;
  out[row * D + t + 256] = y1;
  if (oh) {
    split_bf16(y0, oh[row * D + t],       ol[row * D + t]);
    split_bf16(y1, oh[row * D + t + 256], ol[row * D + t + 256]);
  }
}

// ---------------------------------------------------------------------------
// Launch
// ---------------------------------------------------------------------------
extern "C" void kernel_launch(void* const* d_in, const int* in_sizes, int n_in,
                              void* d_out, int out_size, void* d_ws, size_t ws_size,
                              hipStream_t stream) {
  const float* x    = (const float*)d_in[0];
  const float* Wq   = (const float*)d_in[1];
  const float* bq   = (const float*)d_in[2];
  const float* Wk   = (const float*)d_in[3];
  const float* bk   = (const float*)d_in[4];
  const float* Wv   = (const float*)d_in[5];
  const float* bv   = (const float*)d_in[6];
  const float* Wo   = (const float*)d_in[7];
  const float* bo   = (const float*)d_in[8];
  const float* g1   = (const float*)d_in[9];
  const float* b1   = (const float*)d_in[10];
  const float* Wff1 = (const float*)d_in[11];
  const float* bff1 = (const float*)d_in[12];
  const float* Wff2 = (const float*)d_in[13];
  const float* bff2 = (const float*)d_in[14];
  const float* g2   = (const float*)d_in[15];
  const float* b2   = (const float*)d_in[16];

  const int Bsz = 4, S = 2048, D = 512, H = 8, DFF = 2048;
  const size_t R  = (size_t)Bsz * S;   // 8192 rows
  const size_t RD = R * D;             // 4M
  const size_t DD = (size_t)D * D;     // 256K
  const size_t DF = (size_t)D * DFF;   // 1M

  // Workspace carve (bytes)
  char* p = (char*)d_ws;
  auto alloc_f32  = [&](size_t n) { float*  r = (float*)p;  p += n * 4; return r; };
  auto alloc_bf16 = [&](size_t n) { __bf16* r = (__bf16*)p; p += n * 2; return r; };

  float*  xpe  = alloc_f32(RD);
  __bf16* xpeH = alloc_bf16(RD);
  __bf16* xpeL = alloc_bf16(RD);
  float*  qb_  = alloc_f32(RD);
  float*  kb_  = alloc_f32(RD);
  float*  vb_  = alloc_f32(RD);
  __bf16* ctxH = alloc_bf16(RD);
  __bf16* ctxL = alloc_bf16(RD);
  float*  hbuf = alloc_f32(RD);
  __bf16* hH   = alloc_bf16(RD);
  __bf16* hL   = alloc_bf16(RD);
  __bf16* ff1H = alloc_bf16(R * DFF);
  __bf16* ff1L = alloc_bf16(R * DFF);
  __bf16* WqTH = alloc_bf16(DD); __bf16* WqTL = alloc_bf16(DD);
  __bf16* WkTH = alloc_bf16(DD); __bf16* WkTL = alloc_bf16(DD);
  __bf16* WvTH = alloc_bf16(DD); __bf16* WvTL = alloc_bf16(DD);
  __bf16* WoTH = alloc_bf16(DD); __bf16* WoTL = alloc_bf16(DD);
  __bf16* W1TH = alloc_bf16(DF); __bf16* W1TL = alloc_bf16(DF);
  __bf16* W2TH = alloc_bf16(DF); __bf16* W2TL = alloc_bf16(DF);
  float*  attn_out = qb_;  // reuse (q dead after attention)
  float*  ff2      = kb_;  // reuse (k dead after attention)

  // 1) positional encoding (+ split planes)
  {
    int total = (int)RD;
    pe_add_kernel<<<(total + 255) / 256, 256, 0, stream>>>(x, xpe, xpeH, xpeL, S, D, total);
  }
  // 2) weight prep: transpose + split, once per weight
  {
    int nDD = (int)DD, nDF = (int)DF;
    split_transpose_kernel<<<(nDD + 255) / 256, 256, 0, stream>>>(Wq, WqTH, WqTL, D, D);
    split_transpose_kernel<<<(nDD + 255) / 256, 256, 0, stream>>>(Wk, WkTH, WkTL, D, D);
    split_transpose_kernel<<<(nDD + 255) / 256, 256, 0, stream>>>(Wv, WvTH, WvTL, D, D);
    split_transpose_kernel<<<(nDD + 255) / 256, 256, 0, stream>>>(Wo, WoTH, WoTL, D, D);
    split_transpose_kernel<<<(nDF + 255) / 256, 256, 0, stream>>>(Wff1, W1TH, W1TL, D, DFF);
    split_transpose_kernel<<<(nDF + 255) / 256, 256, 0, stream>>>(Wff2, W2TH, W2TL, DFF, D);
  }
  // 3) Q/K/V projections
  {
    dim3 grid((unsigned)(R / BM), (unsigned)(D / BN));
    gemm_split_kernel<<<grid, 256, 0, stream>>>(xpeH, xpeL, WqTH, WqTL, bq, qb_, nullptr, nullptr, (int)R, D, D, 0);
    gemm_split_kernel<<<grid, 256, 0, stream>>>(xpeH, xpeL, WkTH, WkTL, bk, kb_, nullptr, nullptr, (int)R, D, D, 0);
    gemm_split_kernel<<<grid, 256, 0, stream>>>(xpeH, xpeL, WvTH, WvTL, bv, vb_, nullptr, nullptr, (int)R, D, D, 0);
  }
  // 4) flash attention -> ctx planes
  {
    int waves = Bsz * H * (S / 16);  // 4096
    flash_attn_kernel<<<waves / 4, 128, 0, stream>>>(qb_, kb_, vb_, ctxH, ctxL, Bsz, S, H);
  }
  // 5) output projection
  {
    dim3 grid((unsigned)(R / BM), (unsigned)(D / BN));
    gemm_split_kernel<<<grid, 256, 0, stream>>>(ctxH, ctxL, WoTH, WoTL, bo, attn_out, nullptr, nullptr, (int)R, D, D, 0);
  }
  // 6) h = LN(xpe + attn_out)  (+ split planes)
  add_ln_kernel<<<(unsigned)R, 256, 0, stream>>>(xpe, attn_out, g1, b1, hbuf, hH, hL);
  // 7) FFN
  {
    dim3 grid1((unsigned)(R / BM), (unsigned)(DFF / BN));
    gemm_split_kernel<<<grid1, 256, 0, stream>>>(hH, hL, W1TH, W1TL, bff1, nullptr, ff1H, ff1L, (int)R, D, DFF, 1);
    dim3 grid2((unsigned)(R / BM), (unsigned)(D / BN));
    gemm_split_kernel<<<grid2, 256, 0, stream>>>(ff1H, ff1L, W2TH, W2TL, bff2, ff2, nullptr, nullptr, (int)R, DFF, D, 0);
  }
  // 8) out = LN(h + ff2)
  add_ln_kernel<<<(unsigned)R, 256, 0, stream>>>(hbuf, ff2, g2, b2, (float*)d_out, nullptr, nullptr);
}
